// GNN_76991583748607
// MI455X (gfx1250) — matrix operand order repaired and checked by reading, not compiled, not run on hardware
//
#include <hip/hip_runtime.h>
#include <hip/hip_bf16.h>

typedef __attribute__((ext_vector_type(16))) __bf16    v16bf;
typedef __attribute__((ext_vector_type(8)))  __bf16    v8bf;
typedef __attribute__((ext_vector_type(2)))  __bf16    v2bf;
typedef __attribute__((ext_vector_type(8)))  float     v8f;
typedef __attribute__((ext_vector_type(4)))  float     v4f;
typedef __attribute__((ext_vector_type(4)))  unsigned  v4u;
typedef __attribute__((ext_vector_type(8)))  int       v8i;
typedef __attribute__((ext_vector_type(4)))  int       v4i;

#define NEG_SLOPE 0.2f
#define EPS_F 1e-16f

#define HAS_TDM __has_builtin(__builtin_amdgcn_tensor_load_to_lds)

// Order-preserving float -> uint encoding so segment-max can use
// GLOBAL_ATOMIC_MAX_U32 (no CAS loop).
__device__ __forceinline__ unsigned enc_f(float f) {
  unsigned u = __float_as_uint(f);
  return (u & 0x80000000u) ? ~u : (u | 0x80000000u);
}
__device__ __forceinline__ float dec_f(unsigned u) {
  unsigned v = (u & 0x80000000u) ? (u & 0x7FFFFFFFu) : ~u;
  return __uint_as_float(v);
}

// fp32 -> bf16 streaming conversion, 8 elems/thread (b128 load -> 16B store)
__global__ void gat_cast_bf16(const float* __restrict__ in, __bf16* __restrict__ out,
                              long long n) {
  long long i8 = ((long long)blockIdx.x * blockDim.x + threadIdx.x) * 8;
  const long long stride = (long long)gridDim.x * blockDim.x * 8;
  for (; i8 + 7 < n; i8 += stride) {
    v4f f0 = *(const v4f*)(in + i8);
    v4f f1 = *(const v4f*)(in + i8 + 4);
    v8bf o;
    #pragma unroll
    for (int j = 0; j < 4; ++j) { o[j] = (__bf16)f0[j]; o[4 + j] = (__bf16)f1[j]; }
    *(v8bf*)(out + i8) = o;
  }
  if (blockIdx.x == 0 && threadIdx.x == 0)            // tail (n%8, normally empty)
    for (long long i = n & ~7ll; i < n; ++i) out[i] = (__bf16)in[i];
}

// ---------------------------------------------------------------------------
// GEMM: C[M,Nc] = A[M,K] @ B[K,Nc] ; A bf16 (pre-converted), B fp32 weights
// converted during staging; bf16 WMMA compute, fp32 accumulate.
// Block = 256 threads = 8 waves; block tile 128x64; wave tile 16x64 (4 WMMAs).
// A tile staged by the Tensor Data Mover (wave 0, TDM LDS-padding reproduces
// the 40-bf16 row stride); B tile staged swizzled into fragment-major order.
// Requires Nc % 16 == 0 (true here: 512 / 16).
// ---------------------------------------------------------------------------
#define BM 128
#define BN 64
#define BK 32
#define SA_STRIDE 40   // 32 + 8 pad (bf16) == 16 + 4 DWORDs (TDM pad codes 3/3)

__global__ __launch_bounds__(256) void gat_gemm_wmma(
    const __bf16* __restrict__ A, const float* __restrict__ B,
    float* __restrict__ C, int M, int K, int Nc)
{
  __shared__ __align__(32) __bf16 sA[BM * SA_STRIDE];
  __shared__ __align__(32) __bf16 sBf[(BN / 16) * 32 * 16]; // fragment-major

  const int wave = threadIdx.x >> 5;
  const int lane = threadIdx.x & 31;
  const int row0 = blockIdx.x * BM;
  const int col0 = blockIdx.y * BN;
  const int mloc = lane & 15;
  const int half = lane >> 4;

  v8f acc[4] = {};

  // B staging: thread covers rows (2kp, 2kp+1) x cols (ng*4 .. ng*4+3)
  const int kp = threadIdx.x >> 4;      // 0..15
  const int ng = threadIdx.x & 15;      // 0..15
  const int kk = 2 * kp;
  const int khalf = kk >> 4;
  const int klow  = kk & 15;
  const bool bok  = (col0 + ng * 4) < Nc;   // 4-run never straddles a 16 boundary

#if HAS_TDM
  const unsigned ldsA_off = (unsigned)(unsigned long long)(void*)sA; // AS3 byte offset
#else
  const int arr = threadIdx.x >> 1;         // A row 0..127
  const int acb = (threadIdx.x & 1) * 16;   // A col base 0/16
#endif

  for (int k0 = 0; k0 < K; k0 += BK) {
#if HAS_TDM
    if (threadIdx.x < 32) {
      // ---- D# group 0: count=1, lds_addr, 57-bit global byte addr, type=2
      unsigned long long ga = (unsigned long long)(const void*)A +
                              ((unsigned long long)row0 * K + (unsigned)k0) * 2ull;
      v4u g0;
      g0[0] = 1u;
      g0[1] = ldsA_off;
      g0[2] = (unsigned)(ga & 0xFFFFFFFFull);
      g0[3] = (unsigned)((ga >> 32) & 0x01FFFFFFull) | 0x80000000u;
      // ---- D# group 1: data_size=2B, pad 4 DW every 16 DW (row 32bf16 -> stride 40)
      unsigned tdim0 = (unsigned)(K - k0);   // remaining x elems (OOB -> zero fill)
      unsigned tdim1 = (unsigned)(M - row0); // remaining rows
      v8i g1;
      g1[0] = (int)((1u << 16) | (1u << 20) | (3u << 22) | (3u << 25));
      g1[1] = (int)((tdim0 & 0xFFFFu) << 16);
      g1[2] = (int)((tdim0 >> 16) | ((tdim1 & 0xFFFFu) << 16));
      g1[3] = (int)((tdim1 >> 16) | ((unsigned)BK << 16));       // tile_dim0 = 32
      g1[4] = (int)(unsigned)BM;                                  // tile_dim1 = 128
      g1[5] = (int)(unsigned)K;                                   // dim0 stride
      g1[6] = 0;
      g1[7] = 0;
      v4i gz  = {0, 0, 0, 0};
      v8i gz8 = {0, 0, 0, 0, 0, 0, 0, 0};
      __builtin_amdgcn_tensor_load_to_lds(g0, g1, gz, gz, gz8, 0);
    }
#else
    {
      const bool rok = (row0 + arr) < M;
      const __bf16* Ap = A + (size_t)(row0 + arr) * K + k0 + acb;
      #pragma unroll
      for (int j = 0; j < 16; ++j)
        sA[arr * SA_STRIDE + acb + j] = rok ? Ap[j] : (__bf16)0.f;
    }
#endif

    // stage B (BK x BN) fp32 -> bf16, fragment-major:
    // element (k,n) of col-tile t -> sBf[(t*32 + (k>>4)*16 + (n&15))*16 + (k&15)]
    {
      v4f r0 = {0.f, 0.f, 0.f, 0.f}, r1 = r0;
      if (bok) {
        const float* Bp = B + (size_t)(k0 + kk) * Nc + col0 + ng * 4;
        r0 = *(const v4f*)Bp;
        r1 = *(const v4f*)(Bp + Nc);
      }
      #pragma unroll
      for (int c = 0; c < 4; ++c) {
        int n = ng * 4 + c;
        int frag = ((n >> 4) * 32) + (khalf * 16) + (n & 15);
        v2bf p; p[0] = (__bf16)r0[c]; p[1] = (__bf16)r1[c];
        *(v2bf*)&sBf[(frag << 4) + klow] = p;   // packed b32 store (klow even)
      }
    }

#if HAS_TDM
    if (threadIdx.x < 32) __builtin_amdgcn_s_wait_tensorcnt(0);
#endif
    __syncthreads();

    // A fragment: contiguous k = 16*half + s (matches fragment-major B)
    const __bf16* pa = &sA[(wave * 16 + mloc) * SA_STRIDE + half * 16];
    v8bf a0 = *(const v8bf*)pa;
    v8bf a1 = *(const v8bf*)(pa + 8);
    v16bf av = __builtin_shufflevector(a0, a1, 0, 1, 2, 3, 4, 5, 6, 7,
                                       8, 9, 10, 11, 12, 13, 14, 15);
    // pull all four B fragments first so the WMMAs issue back-to-back
    v16bf bv[4];
    #pragma unroll
    for (int t = 0; t < 4; ++t)
      bv[t] = *(const v16bf*)&sBf[((t * 32) + lane) << 4];
    #pragma unroll
    for (int t = 0; t < 4; ++t)
      acc[t] = __builtin_amdgcn_wmma_f32_16x16x32_bf16(
          false, av, false, bv[t], (short)0, acc[t], false, false);
    __syncthreads();
  }

  // store: 32-bit C/D 16x16 layout: lane&15 -> N, VGPR r -> M=r (+8 for half 1)
  #pragma unroll
  for (int t = 0; t < 4; ++t) {
    int col = col0 + t * 16 + mloc;
    if (col >= Nc) continue;
    #pragma unroll
    for (int r = 0; r < 8; ++r) {
      int row = row0 + wave * 16 + half * 8 + r;
      if (row < M) C[(size_t)row * Nc + col] = acc[t][r];
    }
  }
}

// ---------------------------------------------------------------------------
// Attention-logit dot products: als[n,h] = <xs[n,h,:], a_s[h,:]>, same for ald
// ---------------------------------------------------------------------------
__global__ void gat_al(const float* __restrict__ xs, const float* __restrict__ xd,
                       const float* __restrict__ as_, const float* __restrict__ ad_,
                       float* __restrict__ als, float* __restrict__ ald,
                       int H, int C)
{
  __shared__ float red[128];
  const int n = blockIdx.x;
  const int t = threadIdx.x;
  for (int h = 0; h < H; ++h) {
    float v1 = 0.f, v2 = 0.f;
    if (t < C) {
      size_t base = (size_t)n * H * C + (size_t)h * C + t;
      v1 = xs[base] * as_[h * C + t];
      v2 = xd[base] * ad_[h * C + t];
    }
    red[t] = v1; __syncthreads();
    for (int s = 64; s > 0; s >>= 1) { if (t < s) red[t] += red[t + s]; __syncthreads(); }
    if (t == 0) als[(size_t)n * H + h] = red[0];
    __syncthreads();
    red[t] = v2; __syncthreads();
    for (int s = 64; s > 0; s >>= 1) { if (t < s) red[t] += red[t + s]; __syncthreads(); }
    if (t == 0) ald[(size_t)n * H + h] = red[0];
    __syncthreads();
  }
}

// Edge pass 1: e = leaky_relu(als[src]+ald[dst]); segment max via u32 atomicMax
__global__ void gat_edge_logits_max(const int* __restrict__ src, const int* __restrict__ dst,
                                    const float* __restrict__ als, const float* __restrict__ ald,
                                    float* __restrict__ eb, unsigned* __restrict__ menc,
                                    int E, int H)
{
  long long tid = (long long)blockIdx.x * blockDim.x + threadIdx.x;
  if (tid >= (long long)E * H) return;
  int h = (int)(tid % H);
  int e = (int)(tid / H);
  float v = als[(size_t)src[e] * H + h] + ald[(size_t)dst[e] * H + h];
  float lr = v > 0.f ? v : NEG_SLOPE * v;
  eb[tid] = lr;
  atomicMax(&menc[(size_t)dst[e] * H + h], enc_f(lr));
}

// Edge pass 2: ex = exp(e - m[dst]); segment sum
__global__ void gat_edge_exp_sum(const int* __restrict__ dst, float* __restrict__ eb,
                                 const unsigned* __restrict__ menc, float* __restrict__ den,
                                 int E, int H)
{
  long long tid = (long long)blockIdx.x * blockDim.x + threadIdx.x;
  if (tid >= (long long)E * H) return;
  int h = (int)(tid % H);
  int e = (int)(tid / H);
  size_t di = (size_t)dst[e] * H + h;
  float ex = __expf(eb[tid] - dec_f(menc[di]));
  eb[tid] = ex;
  atomicAdd(&den[di], ex);
}

// Edge pass 3: alpha = ex / (den[dst] + eps)
__global__ void gat_edge_alpha(const int* __restrict__ dst, float* __restrict__ eb,
                               const float* __restrict__ den, int E, int H)
{
  long long tid = (long long)blockIdx.x * blockDim.x + threadIdx.x;
  if (tid >= (long long)E * H) return;
  int h = (int)(tid % H);
  int e = (int)(tid / H);
  eb[tid] = eb[tid] / (den[(size_t)dst[e] * H + h] + EPS_F);
}

// Edge pass 4: out[dst] += xs[src] * alpha ; 4 elems per thread (b128 gather)
__global__ void gat_edge_scatter(const int* __restrict__ src, const int* __restrict__ dst,
                                 const float* __restrict__ xs, const float* __restrict__ alpha,
                                 float* __restrict__ out, int E, int H, int C)
{
  const int HC = H * C;
  const int HC4 = HC >> 2;
  long long tid = (long long)blockIdx.x * blockDim.x + threadIdx.x;
  if (tid >= (long long)E * HC4) return;
  int e = (int)(tid / HC4);
  int idx = ((int)(tid % HC4)) << 2;      // C % 4 == 0, run stays in one head
  int h = idx / C;
  float a = alpha[(size_t)e * H + h];
  const v4f xv = *(const v4f*)&xs[(size_t)src[e] * HC + idx];
  float* op = &out[(size_t)dst[e] * HC + idx];
  atomicAdd(op + 0, xv[0] * a);
  atomicAdd(op + 1, xv[1] * a);
  atomicAdd(op + 2, xv[2] * a);
  atomicAdd(op + 3, xv[3] * a);
}

// Epilogue: out = acc + bias (optional ReLU)
__global__ void gat_bias_act(const float* __restrict__ acc, const float* __restrict__ b,
                             float* __restrict__ out, int N, int HC, int relu)
{
  long long tid = (long long)blockIdx.x * blockDim.x + threadIdx.x;
  if (tid >= (long long)N * HC) return;
  int i = (int)(tid % HC);
  float v = acc[tid] + b[i];
  out[tid] = relu ? fmaxf(v, 0.f) : v;
}

// ---------------------------------------------------------------------------
static inline int cdiv_ll(long long a, long long b) { return (int)((a + b - 1) / b); }

extern "C" void kernel_launch(void* const* d_in, const int* in_sizes, int n_in,
                              void* d_out, int out_size, void* d_ws, size_t ws_size,
                              hipStream_t stream)
{
  (void)n_in; (void)out_size; (void)ws_size;
  const float* x   = (const float*)d_in[0];
  const int*   ei  = (const int*)d_in[1];
  const float* Ws1 = (const float*)d_in[2];
  const float* Wd1 = (const float*)d_in[3];
  const float* as1 = (const float*)d_in[4];
  const float* ad1 = (const float*)d_in[5];
  const float* b1  = (const float*)d_in[6];
  const float* Ws2 = (const float*)d_in[7];
  const float* Wd2 = (const float*)d_in[8];
  const float* as2 = (const float*)d_in[9];
  const float* ad2 = (const float*)d_in[10];
  const float* b2  = (const float*)d_in[11];
  const float* Ws3 = (const float*)d_in[12];
  const float* Wd3 = (const float*)d_in[13];
  const float* as3 = (const float*)d_in[14];
  const float* ad3 = (const float*)d_in[15];
  const float* b3  = (const float*)d_in[16];

  const int F_in = 128;
  const int N = in_sizes[0] / F_in;
  const int E = in_sizes[1] / 2;
  const int* src = ei;
  const int* dst = ei + E;

  // workspace layout
  const size_t HCmax = 512;
  float*    xs   = (float*)d_ws;                     // N*512 : src projection
  float*    xd   = xs + (size_t)N * HCmax;           // N*512 : dst proj, then msg accumulator
  float*    hb   = xd + (size_t)N * HCmax;           // N*512 : layer activations
  float*    eb   = hb + (size_t)N * HCmax;           // E*4   : e / ex / alpha
  float*    den  = eb + (size_t)E * 4;               // N*4
  unsigned* menc = (unsigned*)(den + (size_t)N * 4); // N*4
  float*    als  = (float*)(menc + (size_t)N * 4);   // N*4
  float*    ald  = als + (size_t)N * 4;              // N*4
  __bf16*   Abf  = (__bf16*)(ald + (size_t)N * 4);   // N*512 bf16 : TDM-ready activations

  auto layer = [&](const float* in, int K, const float* Ws, const float* Wd,
                   const float* a_s, const float* a_d, const float* bias,
                   int H, int C, float* outp, bool relu)
  {
    const int HC = H * C;
    // activations -> bf16 once per layer (shared by both projections)
    const long long nA = (long long)N * K;
    int cgrid = cdiv_ll(nA / 8, 256); if (cgrid > 65535) cgrid = 65535;
    gat_cast_bf16<<<cgrid, 256, 0, stream>>>(in, Abf, nA);

    dim3 gblk(256);
    dim3 ggrid(cdiv_ll(N, BM), cdiv_ll(HC, BN));
    gat_gemm_wmma<<<ggrid, gblk, 0, stream>>>(Abf, Ws, xs, N, K, HC);
    gat_gemm_wmma<<<ggrid, gblk, 0, stream>>>(Abf, Wd, xd, N, K, HC);
    gat_al<<<N, 128, 0, stream>>>(xs, xd, a_s, a_d, als, ald, H, C);

    (void)hipMemsetAsync(menc, 0, (size_t)N * H * sizeof(unsigned), stream); // enc(-inf) ~ 0
    (void)hipMemsetAsync(den,  0, (size_t)N * H * sizeof(float), stream);

    const long long EH = (long long)E * H;
    gat_edge_logits_max<<<cdiv_ll(EH, 256), 256, 0, stream>>>(src, dst, als, ald, eb, menc, E, H);
    gat_edge_exp_sum  <<<cdiv_ll(EH, 256), 256, 0, stream>>>(dst, eb, menc, den, E, H);
    gat_edge_alpha    <<<cdiv_ll(EH, 256), 256, 0, stream>>>(dst, eb, den, E, H);

    // xd is dead (al_d consumed) -> reuse as message accumulator
    (void)hipMemsetAsync(xd, 0, (size_t)N * HC * sizeof(float), stream);
    const long long EHC4 = (long long)E * (HC >> 2);
    gat_edge_scatter<<<cdiv_ll(EHC4, 256), 256, 0, stream>>>(src, dst, xs, eb, xd, E, H, C);

    const long long NHC = (long long)N * HC;
    gat_bias_act<<<cdiv_ll(NHC, 256), 256, 0, stream>>>(xd, bias, outp, N, HC, relu ? 1 : 0);
  };

  // layer 1: 128 -> 4x128 concat, ReLU
  layer(x,  128, Ws1, Wd1, as1, ad1, b1, 4, 128, hb, true);
  // layer 2: 512 -> 4x128 concat, ReLU
  layer(hb, 512, Ws2, Wd2, as2, ad2, b2, 4, 128, hb, true);
  // layer 3: 512 -> 1x16, mean over 1 head == identity, no ReLU
  layer(hb, 512, Ws3, Wd3, as3, ad3, b3, 1, 16, (float*)d_out, false);
}
